// ViT_Toy_73512660238392
// MI455X (gfx1250) — matrix operand order repaired and checked by reading, not compiled, not run on hardware
//
#include <hip/hip_runtime.h>
#include <hip/hip_bf16.h>

// ---------------------------------------------------------------------------
// ViT-Base forward for MI455X (gfx1250, wave32, WMMA).
// GEMM: v_wmma_f32_16x16x32_f16, 2x2 register blocking per wave, 64x128 block
// tile, ping-pong LDS double buffering fed by GLOBAL_LOAD_ASYNC_TO_LDS_B128
// (ASYNCcnt) so HBM latency overlaps matrix work. All B operands are
// pre-transposed to [N][K] so every global load is contiguous b128.
// LN / softmax / GELU stay fp32.
// ---------------------------------------------------------------------------

typedef __attribute__((ext_vector_type(16))) _Float16 v16h;
typedef __attribute__((ext_vector_type(8)))  float    v8f;

#define VB      8
#define VC      3
#define VIMG    384
#define VP      16
#define VDIM    768
#define VHEADS  12
#define VHD     64
#define VMLP    3072
#define VDEPTH  12
#define VNCLS   1000
#define VG      (VIMG / VP)          // 24
#define VNP     (VG * VG)            // 576
#define VSEQ    (VNP + 1)            // 577
#define VSEQP   608                  // 577 padded to mult of 32
#define VROWS   (VB * VSEQ)          // 4616

#define TM 64
#define TN 128
#define KC 32
#define LDSP (KC + 8)                      // 40 halves = 80B row pitch (16B mult)
#define SMEM_HALVES ((TM + TN) * LDSP)     // one ping-pong buffer
#define TRASH_OFS   (2 * SMEM_HALVES * 2)  // byte offset of 64B trash slot

// ---------------------------------------------------------------------------
// Batched f16 WMMA GEMM:  C = alpha * A @ B + bias, optional GELU.
// A : row-major [M x K], lda == K, per-batch stride sA.
// Bw: row-major [N x K] (B transposed), ldb == K, per-batch stride sB.
// C : fp32 (Cf) and/or f16 (Ch), leading dim ldc, per-batch stride sC.
// ---------------------------------------------------------------------------
__global__ __launch_bounds__(256)
void gemm_wmma_f16(const _Float16* __restrict__ A,
                   const _Float16* __restrict__ Bw,
                   const float*    __restrict__ bias,
                   float*          __restrict__ Cf,
                   _Float16*       __restrict__ Ch,
                   int M, int N, int K, int ldc,
                   long sA, long sB, long sC,
                   int gelu, float alpha)
{
    // single shared object -> LDS offset 0; raw offsets used by async asm
    __shared__ __align__(16) _Float16 smem[2 * SMEM_HALVES + 32];

#define AS(b, r, c) smem[(b) * SMEM_HALVES + (r) * LDSP + (c)]
#define BT(b, r, c) smem[(b) * SMEM_HALVES + TM * LDSP + (r) * LDSP + (c)]

    const int tid  = threadIdx.x;
    const int lane = tid & 31;
    const int wid  = tid >> 5;          // 0..7
    const int mi   = wid & 1;           // 2 M-subtiles of 32
    const int ni   = wid >> 1;          // 4 N-subtiles of 32
    const int z    = blockIdx.z;
    const int m0   = blockIdx.y * TM;
    const int n0   = blockIdx.x * TN;

    A  += (long)z * sA;
    Bw += (long)z * sB;

    v8f acc[2][2] = {};

    const int arow = tid >> 2;          // 0..63
    const int acol = (tid & 3) * 8;     // 0,8,16,24   (16B units)
    const int brow = tid >> 1;          // 0..127
    const int bcol = (tid & 1) * 16;    // 0,16        (32B units)

    const int gm = m0 + arow;
    const int gn = n0 + brow;
    const int aValid = (gm < M);
    const int bValid = (gn < N);

    // clamped global sources (m0/n0 are always in range)
    const _Float16* aptr = A  + (long)(aValid ? gm : m0) * K + acol;
    const _Float16* bptr = Bw + (long)(bValid ? gn : n0) * K + bcol;

    // LDS byte offsets for async destinations (assumes smem at LDS offset 0)
    const unsigned aofs = aValid ? (unsigned)((arow * LDSP + acol) * 2)
                                 : (unsigned)TRASH_OFS;
    const unsigned bofs = bValid ? (unsigned)(((TM + brow) * LDSP + bcol) * 2)
                                 : (unsigned)TRASH_OFS;
    const unsigned bufB = (unsigned)(SMEM_HALVES * 2);

    // pad rows are never staged: zero them once in both buffers
    if (!aValid) {
#pragma unroll
        for (int j = 0; j < 8; ++j) {
            AS(0, arow, acol + j) = (_Float16)0.f;
            AS(1, arow, acol + j) = (_Float16)0.f;
        }
    }
    if (!bValid) {
#pragma unroll
        for (int j = 0; j < 16; ++j) {
            BT(0, brow, bcol + j) = (_Float16)0.f;
            BT(1, brow, bcol + j) = (_Float16)0.f;
        }
    }

    // async stage of one K-chunk into buffer `buf` (3 ops per wave, uniform)
    auto stage = [&](int buf, int k0) {
        const unsigned la = aValid ? (aofs + (unsigned)buf * bufB) : aofs;
        const _Float16* sa = aptr + k0;
        asm volatile("global_load_async_to_lds_b128 %0, %1, off"
                     :: "v"(la), "v"(sa) : "memory");
        const unsigned lb = bValid ? (bofs + (unsigned)buf * bufB) : bofs;
        const _Float16* sb = bptr + k0;
        asm volatile("global_load_async_to_lds_b128 %0, %1, off"
                     :: "v"(lb), "v"(sb) : "memory");
        const unsigned lb2 = lb + 16;
        asm volatile("global_load_async_to_lds_b128 %0, %1, off"
                     :: "v"(lb2), "v"(sb + 8) : "memory");
    };

    stage(0, 0);
    int cur = 0;

    for (int k0 = 0; k0 < K; k0 += KC) {
        if (k0 + KC < K) {
            stage(cur ^ 1, k0 + KC);            // prefetch next chunk (async)
            asm volatile("s_wait_asynccnt 0x3" ::: "memory");  // cur chunk done
        } else {
            asm volatile("s_wait_asynccnt 0x0" ::: "memory");
        }
        __syncthreads();

        // ---- fragments (ISA 16-bit striped layouts) ----
        union frag { v16h v; unsigned int u[8]; };
        frag a[2], b[2];
        const int mr = lane & 15;
        const int hi = lane >> 4;
#pragma unroll
        for (int r = 0; r < 8; ++r) {
            const int ka = 2 * r + (r >= 4 ? 8 : 0) + 8 * hi;   // A K-pair
            const int kb = 2 * r + (r >= 4 ? 8 : 0) + 8 * hi;   // B K-pair
            a[0].u[r] = *(const unsigned int*)&AS(cur, mi * 32 + mr, ka);
            a[1].u[r] = *(const unsigned int*)&AS(cur, mi * 32 + 16 + mr, ka);
            b[0].u[r] = *(const unsigned int*)&BT(cur, ni * 32 + mr, kb);
            b[1].u[r] = *(const unsigned int*)&BT(cur, ni * 32 + 16 + mr, kb);
        }
#pragma unroll
        for (int ti = 0; ti < 2; ++ti)
#pragma unroll
            for (int tj = 0; tj < 2; ++tj)
                acc[ti][tj] = __builtin_amdgcn_wmma_f32_16x16x32_f16(
                    false, a[ti].v, false, b[tj].v, (short)0, acc[ti][tj],
                    false, false);
        __syncthreads();   // protect cur buffer before it is re-staged
        cur ^= 1;
    }

    // ---- epilogue ----
    const int hi = lane >> 4;
#pragma unroll
    for (int tj = 0; tj < 2; ++tj) {
        const int col = n0 + ni * 32 + tj * 16 + (lane & 15);
        if (col >= N) continue;
        const float bv = bias ? bias[col] : 0.f;
#pragma unroll
        for (int ti = 0; ti < 2; ++ti) {
#pragma unroll
            for (int r = 0; r < 8; ++r) {
                const int row = m0 + mi * 32 + ti * 16 + r + 8 * hi;
                if (row < M) {
                    float val = alpha * acc[ti][tj][r] + bv;
                    if (gelu) val = 0.5f * val * (1.f + erff(val * 0.70710678118f));
                    const long off = (long)z * sC + (long)row * ldc + col;
                    if (Cf) Cf[off] = val;
                    if (Ch) Ch[off] = (_Float16)val;
                }
            }
        }
    }
#undef AS
#undef BT
}

// ---------------------------------------------------------------------------
// Elementwise / reduction helpers
// ---------------------------------------------------------------------------

// src fp32 [D][K][N] -> dst f16 [D][N][K]  (transpose + convert)
__global__ void convert_transpose_f16(const float* __restrict__ src,
                                      _Float16* __restrict__ dst,
                                      int D, int K, int N)
{
    long idx = (long)blockIdx.x * blockDim.x + threadIdx.x;
    const long kn = (long)K * N;
    const long n = (long)D * kn;
    if (idx >= n) return;
    const int d  = (int)(idx / kn);
    const long r = idx % kn;
    const int nn = (int)(r / K);
    const int kk = (int)(r % K);
    dst[idx] = (_Float16)src[(long)d * kn + (long)kk * N + nn];
}

// x[B,C,384,384] -> patches f16 [B*576, 768] (feature = (py*16+px)*3 + c)
__global__ void im2col_kernel(const float* __restrict__ x,
                              _Float16* __restrict__ out)
{
    long idx = (long)blockIdx.x * blockDim.x + threadIdx.x;
    const long n = (long)VB * VNP * (VP * VP * VC);
    if (idx >= n) return;
    int f = (int)(idx % 768);
    long t = idx / 768;
    int p = (int)(t % VNP);
    int b = (int)(t / VNP);
    int c  = f % VC;
    int px = (f / VC) % VP;
    int py = f / (VC * VP);
    int gx = p % VG;
    int gy = p / VG;
    long src = (((long)b * VC + c) * VIMG + (gy * VP + py)) * VIMG + (gx * VP + px);
    out[idx] = (_Float16)x[src];
}

// tok[b][0] = cls + pos[0]; tok[b][i] += pos[i]; emit f16 copy.
__global__ void add_pos_cls(float* __restrict__ tok,
                            const float* __restrict__ pos,
                            const float* __restrict__ cls,
                            _Float16* __restrict__ t16)
{
    long idx = (long)blockIdx.x * blockDim.x + threadIdx.x;
    const long n = (long)VB * VSEQ * VDIM;
    if (idx >= n) return;
    int c = (int)(idx % VDIM);
    long t = idx / VDIM;
    int i = (int)(t % VSEQ);
    float v;
    if (i == 0) v = cls[c] + pos[c];
    else        v = tok[idx] + pos[(long)i * VDIM + c];
    tok[idx] = v;
    t16[idx] = (_Float16)v;
}

// qkv fp32 [B,577,2304] -> q16/k16 [B*H, 608, 64], vT16 [B*H, 64, 608]
__global__ void repack_qkv(const float* __restrict__ qkv,
                           _Float16* __restrict__ q16,
                           _Float16* __restrict__ k16,
                           _Float16* __restrict__ vT16)
{
    long idx = (long)blockIdx.x * blockDim.x + threadIdx.x;
    const long n = (long)VB * VHEADS * VSEQP * VHD;
    if (idx >= n) return;
    int hd = (int)(idx & 63);
    long t = idx >> 6;
    int h = (int)(t % VHEADS); t /= VHEADS;
    int i = (int)(t % VSEQP);
    int b = (int)(t / VSEQP);
    const long bh  = (long)b * VHEADS + h;
    const long dst = (bh * VSEQP + i) * VHD + hd;
    const long dvT = (bh * VHD + hd) * VSEQP + i;
    if (i < VSEQ) {
        long src = ((long)b * VSEQ + i) * (3 * VDIM) + h * VHD + hd;
        q16[dst] = (_Float16)qkv[src];
        k16[dst] = (_Float16)qkv[src + VDIM];
        vT16[dvT] = (_Float16)qkv[src + 2 * VDIM];
    } else {
        q16[dst] = (_Float16)0.f;
        k16[dst] = (_Float16)0.f;
        vT16[dvT] = (_Float16)0.f;
    }
}

// Wave-per-row softmax over 577 valid cols, f16 output padded to 608.
__global__ void softmax_rows(const float* __restrict__ S,
                             _Float16* __restrict__ P)
{
    const long base = (long)blockIdx.x * VSEQP;
    const int lane = threadIdx.x;
    float mx = -1e30f;
    for (int j = lane; j < VSEQ; j += 32) mx = fmaxf(mx, S[base + j]);
#pragma unroll
    for (int m = 16; m; m >>= 1) mx = fmaxf(mx, __shfl_xor(mx, m, 32));
    float sum = 0.f;
    for (int j = lane; j < VSEQ; j += 32) sum += __expf(S[base + j] - mx);
#pragma unroll
    for (int m = 16; m; m >>= 1) sum += __shfl_xor(sum, m, 32);
    const float inv = 1.f / sum;
    for (int j = lane; j < VSEQP; j += 32)
        P[base + j] = (j < VSEQ) ? (_Float16)(__expf(S[base + j] - mx) * inv)
                                 : (_Float16)0.f;
}

// attn-out f16 [B*H, 608, 64] -> o16 f16 [B*577, 768]
__global__ void repack_attnout(const _Float16* __restrict__ och,
                               _Float16* __restrict__ o16)
{
    long idx = (long)blockIdx.x * blockDim.x + threadIdx.x;
    const long n = (long)VB * VHEADS * VSEQ * VHD;
    if (idx >= n) return;
    int hd = (int)(idx & 63);
    long t = idx >> 6;
    int h = (int)(t % VHEADS); t /= VHEADS;
    int i = (int)(t % VSEQ);
    int b = (int)(t / VSEQ);
    o16[((long)b * VSEQ + i) * VDIM + h * VHD + hd] =
        och[(((long)b * VHEADS + h) * VSEQP + i) * VHD + hd];
}

// Wave-per-row:  t = LN(t + add) * g + b ; emit fp32 state + f16 copy.
__global__ void ln_residual(float* __restrict__ t,
                            const float* __restrict__ add,
                            const float* __restrict__ g,
                            const float* __restrict__ bb,
                            _Float16* __restrict__ t16)
{
    const long base = (long)blockIdx.x * VDIM;
    const int lane = threadIdx.x;
    float xs[VDIM / 32];
    float s = 0.f;
#pragma unroll
    for (int u = 0; u < VDIM / 32; ++u) {
        const int c = lane + u * 32;
        const float x = t[base + c] + add[base + c];
        xs[u] = x; s += x;
    }
#pragma unroll
    for (int m = 16; m; m >>= 1) s += __shfl_xor(s, m, 32);
    const float mean = s * (1.f / VDIM);
    float v = 0.f;
#pragma unroll
    for (int u = 0; u < VDIM / 32; ++u) { const float d = xs[u] - mean; v += d * d; }
#pragma unroll
    for (int m = 16; m; m >>= 1) v += __shfl_xor(v, m, 32);
    const float inv = rsqrtf(v * (1.f / VDIM) + 1e-5f);
#pragma unroll
    for (int u = 0; u < VDIM / 32; ++u) {
        const int c = lane + u * 32;
        const float y = g[c] * (xs[u] - mean) * inv + bb[c];
        t[base + c] = y;
        t16[base + c] = (_Float16)y;
    }
}

// ---------------------------------------------------------------------------
// Host orchestration
// ---------------------------------------------------------------------------
extern "C" void kernel_launch(void* const* d_in, const int* in_sizes, int n_in,
                              void* d_out, int out_size, void* d_ws, size_t ws_size,
                              hipStream_t stream)
{
    (void)in_sizes; (void)n_in; (void)out_size; (void)ws_size;

    const float* x       = (const float*)d_in[0];
    const float* proj_w  = (const float*)d_in[1];
    const float* proj_b  = (const float*)d_in[2];
    const float* pos_emb = (const float*)d_in[3];
    const float* cls_tok = (const float*)d_in[4];
    const float* qkv_w   = (const float*)d_in[5];
    const float* out_w   = (const float*)d_in[6];
    const float* out_b   = (const float*)d_in[7];
    const float* ln1_g   = (const float*)d_in[8];
    const float* ln1_b   = (const float*)d_in[9];
    const float* ff_w1   = (const float*)d_in[10];
    const float* ff_b1   = (const float*)d_in[11];
    const float* ff_w2   = (const float*)d_in[12];
    const float* ff_b2   = (const float*)d_in[13];
    const float* ln2_g   = (const float*)d_in[14];
    const float* ln2_b   = (const float*)d_in[15];
    const float* head_w1 = (const float*)d_in[16];
    const float* head_b1 = (const float*)d_in[17];
    const float* head_w2 = (const float*)d_in[18];
    const float* head_b2 = (const float*)d_in[19];

    // ---- carve workspace ----
    char* wp = (char*)d_ws;
    auto alloc = [&](size_t bytes) -> void* {
        void* r = (void*)wp;
        wp += (bytes + 255) & ~(size_t)255;
        return r;
    };
    const long nQKVW = (long)VDEPTH * VDIM * 3 * VDIM;
    const long nOUTW = (long)VDEPTH * VDIM * VDIM;
    const long nFF1W = (long)VDEPTH * VDIM * VMLP;
    const long nFF2W = (long)VDEPTH * VMLP * VDIM;
    const long nPRJW = (long)VP * VP * VC * VDIM;
    const long nHD1W = (long)VDIM * VMLP;
    const long nHD2W = (long)VMLP * VNCLS;

    _Float16* wqkv16 = (_Float16*)alloc(nQKVW * 2);   // [l][2304][768]
    _Float16* wout16 = (_Float16*)alloc(nOUTW * 2);   // [l][768][768]
    _Float16* wff116 = (_Float16*)alloc(nFF1W * 2);   // [l][3072][768]
    _Float16* wff216 = (_Float16*)alloc(nFF2W * 2);   // [l][768][3072]
    _Float16* wprj16 = (_Float16*)alloc(nPRJW * 2);   // [768][768]
    _Float16* whd116 = (_Float16*)alloc(nHD1W * 2);   // [3072][768]
    _Float16* whd216 = (_Float16*)alloc(nHD2W * 2);   // [1000][3072]

    _Float16* patches16 = (_Float16*)alloc((long)VB * VNP * VDIM * 2);
    float*    tokf      = (float*)   alloc((long)VROWS * VDIM * 4);
    _Float16* t16       = (_Float16*)alloc((long)VROWS * VDIM * 2);
    float*    qkvf      = (float*)   alloc((long)VROWS * 3 * VDIM * 4);
    _Float16* q16       = (_Float16*)alloc((long)VB * VHEADS * VSEQP * VHD * 2);
    _Float16* k16       = (_Float16*)alloc((long)VB * VHEADS * VSEQP * VHD * 2);
    _Float16* vT16      = (_Float16*)alloc((long)VB * VHEADS * VHD * VSEQP * 2);
    float*    scoresf   = (float*)   alloc((long)VB * VHEADS * VSEQ * VSEQP * 4);
    _Float16* p16       = (_Float16*)alloc((long)VB * VHEADS * VSEQ * VSEQP * 2);
    _Float16* och16     = (_Float16*)alloc((long)VB * VHEADS * VSEQP * VHD * 2);
    _Float16* o16       = (_Float16*)alloc((long)VROWS * VDIM * 2);
    float*    obuf      = (float*)   alloc((long)VROWS * VDIM * 4);
    _Float16* h16       = (_Float16*)alloc((long)VROWS * VMLP * 2);
    _Float16* hcls16    = (_Float16*)alloc((long)VB * VMLP * 2);

    auto convT = [&](const float* s, _Float16* d, int D, int K, int N) {
        const long n = (long)D * K * N;
        convert_transpose_f16<<<(int)((n + 255) / 256), 256, 0, stream>>>(s, d, D, K, N);
    };
    auto gemm = [&](const _Float16* A, const _Float16* Bw, const float* bias,
                    float* Cf, _Float16* Ch, int M, int N, int K, int ldc,
                    long sA, long sB, long sC, int nz, int gelu, float alpha) {
        dim3 grid((N + TN - 1) / TN, (M + TM - 1) / TM, nz);
        gemm_wmma_f16<<<grid, 256, 0, stream>>>(A, Bw, bias, Cf, Ch, M, N, K,
                                                ldc, sA, sB, sC, gelu, alpha);
    };

    // ---- weight convert + transpose to [N][K] f16 (deterministic) ----
    convT(qkv_w,   wqkv16, VDEPTH, VDIM, 3 * VDIM);
    convT(out_w,   wout16, VDEPTH, VDIM, VDIM);
    convT(ff_w1,   wff116, VDEPTH, VDIM, VMLP);
    convT(ff_w2,   wff216, VDEPTH, VMLP, VDIM);
    convT(proj_w,  wprj16, 1, VDIM, VDIM);
    convT(head_w1, whd116, 1, VDIM, VMLP);
    convT(head_w2, whd216, 1, VMLP, VNCLS);

    // ---- patch embedding ----
    {
        const long n = (long)VB * VNP * VDIM;
        im2col_kernel<<<(int)((n + 255) / 256), 256, 0, stream>>>(x, patches16);
    }
    gemm(patches16, wprj16, proj_b, tokf + VDIM, nullptr,
         VNP, VDIM, VDIM, VDIM,
         (long)VNP * VDIM, 0, (long)VSEQ * VDIM, VB, 0, 1.f);
    {
        const long n = (long)VB * VSEQ * VDIM;
        add_pos_cls<<<(int)((n + 255) / 256), 256, 0, stream>>>(tokf, pos_emb,
                                                                cls_tok, t16);
    }

    const float scale = 0.125f;  // HD^-0.5
    const int BH = VB * VHEADS;

    // ---- transformer blocks ----
    for (int l = 0; l < VDEPTH; ++l) {
        const _Float16* wq = wqkv16 + (long)l * VDIM * 3 * VDIM;
        const _Float16* wo = wout16 + (long)l * VDIM * VDIM;
        const _Float16* w1 = wff116 + (long)l * VDIM * VMLP;
        const _Float16* w2 = wff216 + (long)l * VMLP * VDIM;

        // qkv = t @ qkv_w
        gemm(t16, wq, nullptr, qkvf, nullptr,
             VROWS, 3 * VDIM, VDIM, 3 * VDIM, 0, 0, 0, 1, 0, 1.f);
        {
            const long n = (long)BH * VSEQP * VHD;
            repack_qkv<<<(int)((n + 255) / 256), 256, 0, stream>>>(qkvf, q16, k16, vT16);
        }
        // scores = scale * Q @ K^T  (B = K in [j][hd] = [N][K] layout)
        gemm(q16, k16, nullptr, scoresf, nullptr,
             VSEQ, VSEQ, VHD, VSEQP,
             (long)VSEQP * VHD, (long)VSEQP * VHD, (long)VSEQ * VSEQP,
             BH, 0, scale);
        softmax_rows<<<BH * VSEQ, 32, 0, stream>>>(scoresf, p16);
        // O = P @ V  (B = V^T in [hd][j] = [N][K] layout)
        gemm(p16, vT16, nullptr, nullptr, och16,
             VSEQ, VHD, VSEQP, VHD,
             (long)VSEQ * VSEQP, (long)VHD * VSEQP, (long)VSEQP * VHD,
             BH, 0, 1.f);
        {
            const long n = (long)BH * VSEQ * VHD;
            repack_attnout<<<(int)((n + 255) / 256), 256, 0, stream>>>(och16, o16);
        }
        // out projection + bias
        gemm(o16, wo, out_b + (long)l * VDIM, obuf, nullptr,
             VROWS, VDIM, VDIM, VDIM, 0, 0, 0, 1, 0, 1.f);
        ln_residual<<<VROWS, 32, 0, stream>>>(tokf, obuf,
                                              ln1_g + (long)l * VDIM,
                                              ln1_b + (long)l * VDIM, t16);
        // h = GELU(t @ w1 + b1)
        gemm(t16, w1, ff_b1 + (long)l * VMLP, nullptr, h16,
             VROWS, VMLP, VDIM, VMLP, 0, 0, 0, 1, 1, 1.f);
        // ff = h @ w2 + b2
        gemm(h16, w2, ff_b2 + (long)l * VDIM, obuf, nullptr,
             VROWS, VDIM, VMLP, VDIM, 0, 0, 0, 1, 0, 1.f);
        ln_residual<<<VROWS, 32, 0, stream>>>(tokf, obuf,
                                              ln2_g + (long)l * VDIM,
                                              ln2_b + (long)l * VDIM, t16);
    }

    // ---- classifier head on cls tokens (row 0 of each batch) ----
    gemm(t16, whd116, head_b1, nullptr, hcls16,
         1, VMLP, VDIM, VMLP,
         (long)VSEQ * VDIM, 0, (long)VMLP, VB, 1, 1.f);
    gemm(hcls16, whd216, head_b2, (float*)d_out, nullptr,
         VB, VNCLS, VMLP, VNCLS, 0, 0, 0, 1, 0, 1.f);
}
